// CausalSelfAttention_78365973282992
// MI455X (gfx1250) — matrix operand order repaired and checked by reading, not compiled, not run on hardware
//
#include <hip/hip_runtime.h>
#include <math.h>

// ---------------------------------------------------------------------------
// CDNA5 (gfx1250) causal self-attention:
//   pass 0: x -> bf16 ; qkv_w, o_w -> bf16 transposed (N-major)
//   pass 1: QKV GEMM (TDM-staged tiles, double-buffered, WMMA bf16)
//   pass 2: causal flash attention (WMMA bf16, online softmax)
//   pass 3: output projection (TDM-staged tiles, WMMA bf16, fp32 out)
// ---------------------------------------------------------------------------

typedef __attribute__((ext_vector_type(16))) __bf16          v16bf;
typedef __attribute__((ext_vector_type(8)))  float           v8f;
typedef __attribute__((ext_vector_type(8)))  unsigned short  v8u;
typedef __attribute__((ext_vector_type(16))) unsigned short  v16u;
typedef __attribute__((ext_vector_type(4)))  unsigned int    u32x4;
typedef __attribute__((ext_vector_type(8)))  int             i32x8;
typedef __attribute__((ext_vector_type(4)))  int             i32x4;

#define SEQ_T   2048
#define DMODEL  1024
#define NHEADS  16
#define HDIM    64

__device__ __forceinline__ unsigned short f2bf(float f) {
  unsigned int u = __builtin_bit_cast(unsigned int, f);
  unsigned int r = u + 0x7FFFu + ((u >> 16) & 1u);   // round-to-nearest-even
  return (unsigned short)(r >> 16);
}

__device__ __forceinline__ v16bf mk16(v8u lo, v8u hi) {
  v16u v = __builtin_shufflevector(lo, hi, 0,1,2,3,4,5,6,7,8,9,10,11,12,13,14,15);
  return __builtin_bit_cast(v16bf, v);
}

__device__ __forceinline__ v8f wmma_bf16(v16bf a, v16bf b, v8f c) {
  return __builtin_amdgcn_wmma_f32_16x16x32_bf16(false, a, false, b, (short)0, c,
                                                 false, false);
}

// ---------------------------------------------------------------------------
// TDM: DMA one 128x32 bf16 tile (row stride = tensor_d0 elems) into LDS with
// 16B padding after each 64B row (=> 40-short LDS row stride). D# layout per
// CDNA5 ISA ch.8: group0 = {count/type, lds_addr, global_addr}, group1 =
// {data_size, pad, tensor dims, tile dims, dim0 stride}. Groups 2/3 zero (2D).
// This toolchain's builtin takes 6 args (extra i32x8 group, then cpol).
// ---------------------------------------------------------------------------
__device__ __forceinline__ void tdm_load_tile_128x32(
    unsigned lds_byte_off, const unsigned short* gsrc,
    unsigned tensor_d0, unsigned tensor_d1)
{
  unsigned long long ga = (unsigned long long)(uintptr_t)gsrc;
  u32x4 g0;
  g0[0] = 1u;                                         // count=1, user mode
  g0[1] = lds_byte_off;                               // lds_addr (bytes)
  g0[2] = (unsigned)(ga & 0xFFFFFFFFu);               // global_addr[31:0]
  g0[3] = (unsigned)((ga >> 32) & 0x1FFFFFFu)         // global_addr[56:32]
        | (2u << 30);                                 // type = 2 ("image")
  i32x8 g1;
  g1[0] = (int)((1u << 16)                            // data_size = 2 bytes
              | (1u << 20)                            // pad_enable
              | (3u << 22)                            // pad_interval: 64B
              | (3u << 25));                          // pad_amount: 4 dwords=16B
  g1[1] = (int)((tensor_d0 & 0xFFFFu) << 16);         // tensor_dim0[15:0]
  g1[2] = (int)(((tensor_d0 >> 16) & 0xFFFFu)
              | ((tensor_d1 & 0xFFFFu) << 16));       // dim0 hi | dim1 lo
  g1[3] = (int)(((tensor_d1 >> 16) & 0xFFFFu)
              | (32u << 16));                         // dim1 hi | tile_dim0=32
  g1[4] = (int)128;                                   // tile_dim1=128, tile_dim2=0
  g1[5] = (int)tensor_d0;                             // tensor_dim0_stride lo
  g1[6] = 0;
  g1[7] = 0;
  const i32x4 z4 = {0, 0, 0, 0};
  const i32x8 z8 = {0, 0, 0, 0, 0, 0, 0, 0};
  __builtin_amdgcn_tensor_load_to_lds(g0, g1, z4, z4, z8, 0);
}

// ---------------------------------------------------------------------------
// Shared GEMM main loop: acc(128x128 tile) += A[128xK] * B^T[128xK]^T
// A, B are bf16, row-major with 1024-elem rows; staged by TDM, double-buffered.
// ---------------------------------------------------------------------------
__device__ __forceinline__ void gemm_tdm_mainloop(
    const unsigned short* __restrict__ Abase,   // + bm row offset applied
    const unsigned short* __restrict__ Bbase,   // + bn row offset applied
    unsigned a_d1, unsigned b_d1,
    unsigned short* smem,                       // 2 * (5120 A + 5120 B) shorts
    v8f acc[4][2], int tid)
{
  const int lane  = tid & 31;
  const int wid   = tid >> 5;
  const int wm    = (wid & 1) * 64;
  const int wn    = (wid >> 1) * 32;
  const int l15   = lane & 15;
  const int lhalf = lane >> 4;
  const unsigned lbase = (unsigned)(unsigned long long)(uintptr_t)smem;

  if (wid == 0) {
    tdm_load_tile_128x32(lbase,          Abase, 1024u, a_d1);
    tdm_load_tile_128x32(lbase + 10240u, Bbase, 1024u, b_d1);
  }
  int cur = 0;
  for (int kk = 0; kk < 1024; kk += 32) {
    const int nxt = cur ^ 1;
    if (wid == 0) {
      if (kk + 32 < 1024) {
        const unsigned nb = lbase + (unsigned)nxt * 20480u;
        tdm_load_tile_128x32(nb,          Abase + kk + 32, 1024u, a_d1);
        tdm_load_tile_128x32(nb + 10240u, Bbase + kk + 32, 1024u, b_d1);
        __builtin_amdgcn_s_wait_tensorcnt(2);   // current pair complete
      } else {
        __builtin_amdgcn_s_wait_tensorcnt(0);
      }
    }
    __syncthreads();

    const unsigned short* lA = smem + cur * 10240;
    const unsigned short* lB = lA + 5120;
    v16bf af[4];
    for (int im = 0; im < 4; im++) {
      const unsigned short* p = lA + (wm + im*16 + l15) * 40 + lhalf * 8;
      af[im] = mk16(*(const v8u*)p, *(const v8u*)(p + 16));
    }
    v16bf bfr[2];
    for (int in = 0; in < 2; in++) {
      const unsigned short* p = lB + (wn + in*16 + l15) * 40 + lhalf * 8;
      bfr[in] = mk16(*(const v8u*)p, *(const v8u*)(p + 16));
    }
    for (int im = 0; im < 4; im++)
      for (int in = 0; in < 2; in++)
        acc[im][in] = wmma_bf16(af[im], bfr[in], acc[im][in]);

    __syncthreads();
    cur = nxt;
  }
}

// ---------------------------------------------------------------------------
// Pass 0a: fp32 -> bf16 elementwise (x)
// ---------------------------------------------------------------------------
__global__ __launch_bounds__(256) void cvt_bf16_kernel(
    const float* __restrict__ in, unsigned short* __restrict__ out)
{
  int i = blockIdx.x * 256 + threadIdx.x;       // one v8u per thread
  const float4 a = *(const float4*)(in + (size_t)i * 8);
  const float4 b = *(const float4*)(in + (size_t)i * 8 + 4);
  v8u pk;
  pk[0] = f2bf(a.x); pk[1] = f2bf(a.y); pk[2] = f2bf(a.z); pk[3] = f2bf(a.w);
  pk[4] = f2bf(b.x); pk[5] = f2bf(b.y); pk[6] = f2bf(b.z); pk[7] = f2bf(b.w);
  *(v8u*)(out + (size_t)i * 8) = pk;
}

// ---------------------------------------------------------------------------
// Pass 0b: weight transpose + convert: in[k][n] fp32 -> out[n][k] bf16
// K = 1024 rows, ncols columns; 64x64 tiles through LDS.
// ---------------------------------------------------------------------------
__global__ __launch_bounds__(256) void transpose_bf16_kernel(
    const float* __restrict__ in, unsigned short* __restrict__ out, int ncols)
{
  __shared__ __align__(16) unsigned short tile[64][72];   // [n][k], padded
  const int kt = blockIdx.x * 64;
  const int nt = blockIdx.y * 64;
  const int tid = threadIdx.x;
  for (int j = 0; j < 4; j++) {
    int i = tid + 256 * j;                 // 1024 float4's
    int k = i >> 4;
    int c = (i & 15) << 2;
    const float4 v = *(const float4*)(in + (size_t)(kt + k) * ncols + nt + c);
    tile[c+0][k] = f2bf(v.x); tile[c+1][k] = f2bf(v.y);
    tile[c+2][k] = f2bf(v.z); tile[c+3][k] = f2bf(v.w);
  }
  __syncthreads();
  for (int j = 0; j < 2; j++) {
    int i  = tid + 256 * j;                // 512 v8u's
    int nl = i >> 3;
    int kc = (i & 7) << 3;
    *(v8u*)(out + (size_t)(nt + nl) * 1024 + kt + kc) = *(const v8u*)&tile[nl][kc];
  }
}

// ---------------------------------------------------------------------------
// Pass 1: qkv = xbf @ wqkvT^T + b ; scatter Q (pre-scaled), K, V^T as bf16
// ---------------------------------------------------------------------------
__global__ __launch_bounds__(256) void qkv_kernel(
    const unsigned short* __restrict__ xbf, const unsigned short* __restrict__ wT,
    const float* __restrict__ bias,
    unsigned short* __restrict__ qws, unsigned short* __restrict__ kws,
    unsigned short* __restrict__ vtws)
{
  __shared__ __align__(16) unsigned short smem[40960 / 2];   // 40 KB
  const int tid   = threadIdx.x;
  const int lane  = tid & 31;
  const int wid   = tid >> 5;
  const int wm    = (wid & 1) * 64;
  const int wn    = (wid >> 1) * 32;
  const int bm    = blockIdx.x * 128;
  const int bn    = blockIdx.y * 128;
  const int l15   = lane & 15;
  const int lhalf = lane >> 4;

  const v8f vzero = {0.f,0.f,0.f,0.f,0.f,0.f,0.f,0.f};
  v8f acc[4][2];
  for (int i = 0; i < 4; i++) for (int j = 0; j < 2; j++) acc[i][j] = vzero;

  gemm_tdm_mainloop(xbf + (size_t)bm * 1024, wT + (size_t)bn * 1024,
                    4096u, 3072u, smem, acc, tid);

  // --- epilogue: section is block-uniform (128 | 1024) ---
  const int sec = bn >> 10;                      // 0=Q, 1=K, 2=V
  const int nn0 = (bn & 1023) + wn;
  const float bias_c[2] = { bias[bn + wn + l15], bias[bn + wn + 16 + l15] };

  if (sec == 2) {
    // V^T: lane holds 8 consecutive t at fixed d -> one 16B store per frag
    for (int im = 0; im < 4; im++)
      for (int in = 0; in < 2; in++) {
        int nn = nn0 + in*16 + l15;
        int h = nn >> 6, d = nn & 63;
        int r0 = bm + wm + im*16 + 8*lhalf;      // 8 consecutive rows
        int b = r0 >> 11, t0 = r0 & 2047;
        v8u pk;
        for (int v = 0; v < 8; v++) pk[v] = f2bf(acc[im][in][v] + bias_c[in]);
        *(v8u*)(vtws + ((size_t)(b*NHEADS + h)*HDIM + d)*SEQ_T + t0) = pk;
      }
  } else {
    unsigned short* dst = sec ? kws : qws;
    const float mul = sec ? 1.0f : 0.125f;       // fold 1/sqrt(Dh) into Q
    for (int im = 0; im < 4; im++)
      for (int in = 0; in < 2; in++) {
        int nn = nn0 + in*16 + l15;
        int h = nn >> 6, d = nn & 63;
        int r0 = bm + wm + im*16 + 8*lhalf;
        int b = r0 >> 11, t0 = r0 & 2047;
        size_t base = ((size_t)(b*NHEADS + h)*SEQ_T + t0)*HDIM + d;
        for (int v = 0; v < 8; v++)
          dst[base + (size_t)v*HDIM] = f2bf((acc[im][in][v] + bias_c[in]) * mul);
      }
  }
}

// ---------------------------------------------------------------------------
// Pass 2: causal flash attention; one block per (b,h, 128 queries),
// one wave per 16 queries, online softmax, kv tiles of 32 keys.
// ---------------------------------------------------------------------------
__global__ __launch_bounds__(256) void attn_kernel(
    const unsigned short* __restrict__ qws, const unsigned short* __restrict__ kws,
    const unsigned short* __restrict__ vtws, unsigned short* __restrict__ aws)
{
  __shared__ __align__(16) unsigned short lds_p[8][16][32];  // per-wave P bounce
  const int tid   = threadIdx.x;
  const int lane  = tid & 31;
  const int wid   = tid >> 5;
  const int l15   = lane & 15;
  const int lhalf = lane >> 4;
  const int bh    = blockIdx.y;
  const int b     = bh >> 4, h = bh & 15;
  const int qbase = blockIdx.x * 128 + wid * 16;

  const unsigned short* Qh  = qws  + (size_t)bh * SEQ_T * HDIM;
  const unsigned short* Kh  = kws  + (size_t)bh * SEQ_T * HDIM;
  const unsigned short* Vth = vtws + (size_t)bh * HDIM  * SEQ_T;

  v16bf aq[2];
  for (int c = 0; c < 2; c++) {
    const unsigned short* p = Qh + (size_t)(qbase + l15) * HDIM + c*32 + lhalf*8;
    aq[c] = mk16(*(const v8u*)p, *(const v8u*)(p + 16));
  }

  const v8f vzero = {0.f,0.f,0.f,0.f,0.f,0.f,0.f,0.f};
  v8f o[4]; for (int f = 0; f < 4; f++) o[f] = vzero;
  float m_run[8], l_run[8];
  for (int v = 0; v < 8; v++) { m_run[v] = -__builtin_inff(); l_run[v] = 0.f; }

  const int nj = (qbase + 16 + 31) >> 5;
  for (int j = 0; j < nj; j++) {
    const int kv0 = j * 32;

    v8f s[2] = {vzero, vzero};
    for (int c = 0; c < 2; c++)
      for (int fn = 0; fn < 2; fn++) {
        const unsigned short* p =
            Kh + (size_t)(kv0 + fn*16 + l15) * HDIM + c*32 + lhalf*8;
        s[fn] = wmma_bf16(aq[c], mk16(*(const v8u*)p, *(const v8u*)(p + 16)), s[fn]);
      }

    if (kv0 + 31 > qbase) {                       // causal mask, diagonal tiles
      for (int fn = 0; fn < 2; fn++)
        for (int v = 0; v < 8; v++) {
          int row = qbase + v + 8*lhalf;
          int col = kv0 + fn*16 + l15;
          if (col > row) s[fn][v] = -__builtin_inff();
        }
    }

    float p0[8], p1[8];
    for (int v = 0; v < 8; v++) {
      float mx = fmaxf(s[0][v], s[1][v]);
      for (int msk = 8; msk >= 1; msk >>= 1)
        mx = fmaxf(mx, __shfl_xor(mx, msk, 32));
      float mn    = fmaxf(m_run[v], mx);
      float alpha = __expf(m_run[v] - mn);
      float e0 = __expf(s[0][v] - mn);
      float e1 = __expf(s[1][v] - mn);
      float rs = e0 + e1;
      for (int msk = 8; msk >= 1; msk >>= 1)
        rs += __shfl_xor(rs, msk, 32);
      l_run[v] = l_run[v] * alpha + rs;
      m_run[v] = mn;
      p0[v] = e0; p1[v] = e1;
      for (int f = 0; f < 4; f++) o[f][v] *= alpha;
    }

    for (int v = 0; v < 8; v++) {                 // P: C-layout -> A-layout
      lds_p[wid][v + 8*lhalf][l15]      = f2bf(p0[v]);
      lds_p[wid][v + 8*lhalf][16 + l15] = f2bf(p1[v]);
    }
    asm volatile("s_wait_dscnt 0x0" ::: "memory");
    v16bf ap;
    {
      const unsigned short* p = &lds_p[wid][l15][lhalf * 8];
      ap = mk16(*(const v8u*)p, *(const v8u*)(p + 16));
    }

    for (int f = 0; f < 4; f++) {
      const unsigned short* p = Vth + (size_t)(f*16 + l15) * SEQ_T + kv0 + lhalf*8;
      o[f] = wmma_bf16(ap, mk16(*(const v8u*)p, *(const v8u*)(p + 16)), o[f]);
    }
  }

  for (int v = 0; v < 8; v++) {
    float inv = 1.0f / l_run[v];
    int t = qbase + v + 8*lhalf;
    for (int f = 0; f < 4; f++)
      aws[((size_t)b * SEQ_T + t) * DMODEL + h*HDIM + f*16 + l15] =
          f2bf(o[f][v] * inv);
  }
}

// ---------------------------------------------------------------------------
// Pass 3: out = attn @ owT^T + o_b  (fp32 output)
// ---------------------------------------------------------------------------
__global__ __launch_bounds__(256) void oproj_kernel(
    const unsigned short* __restrict__ aws, const unsigned short* __restrict__ owT,
    const float* __restrict__ ob, float* __restrict__ out)
{
  __shared__ __align__(16) unsigned short smem[40960 / 2];
  const int tid   = threadIdx.x;
  const int lane  = tid & 31;
  const int wid   = tid >> 5;
  const int wm    = (wid & 1) * 64;
  const int wn    = (wid >> 1) * 32;
  const int bm    = blockIdx.x * 128;
  const int bn    = blockIdx.y * 128;
  const int l15   = lane & 15;
  const int lhalf = lane >> 4;

  const v8f vzero = {0.f,0.f,0.f,0.f,0.f,0.f,0.f,0.f};
  v8f acc[4][2];
  for (int i = 0; i < 4; i++) for (int j = 0; j < 2; j++) acc[i][j] = vzero;

  gemm_tdm_mainloop(aws + (size_t)bm * 1024, owT + (size_t)bn * 1024,
                    4096u, 1024u, smem, acc, tid);

  const float ob_c[2] = { ob[bn + wn + l15], ob[bn + wn + 16 + l15] };
  for (int im = 0; im < 4; im++)
    for (int in = 0; in < 2; in++)
      for (int v = 0; v < 8; v++)
        out[(size_t)(bm + wm + im*16 + v + 8*lhalf) * DMODEL
            + bn + wn + in*16 + l15] = acc[im][in][v] + ob_c[in];
}

// ---------------------------------------------------------------------------
extern "C" void kernel_launch(void* const* d_in, const int* in_sizes, int n_in,
                              void* d_out, int out_size, void* d_ws, size_t ws_size,
                              hipStream_t stream) {
  (void)in_sizes; (void)n_in; (void)out_size; (void)ws_size;
  const float* x     = (const float*)d_in[0];
  const float* qkv_w = (const float*)d_in[1];
  const float* qkv_b = (const float*)d_in[2];
  const float* o_w   = (const float*)d_in[3];
  const float* o_b   = (const float*)d_in[4];

  unsigned short* xbf   = (unsigned short*)d_ws;            // 4096x1024
  unsigned short* wqkvT = xbf   + 4194304;                  // 3072x1024
  unsigned short* owT   = wqkvT + 3145728;                  // 1024x1024
  unsigned short* qws   = owT   + 1048576;                  // [B,H,T,Dh]
  unsigned short* kws   = qws   + 4194304;                  // [B,H,T,Dh]
  unsigned short* vtws  = kws   + 4194304;                  // [B,H,Dh,T]
  unsigned short* aws   = vtws  + 4194304;                  // [B,T,C]  (48MB total)

  cvt_bf16_kernel      <<<2048,          256, 0, stream>>>(x, xbf);
  transpose_bf16_kernel<<<dim3(16, 48),  256, 0, stream>>>(qkv_w, wqkvT, 3072);
  transpose_bf16_kernel<<<dim3(16, 16),  256, 0, stream>>>(o_w, owT, 1024);
  qkv_kernel           <<<dim3(32, 24),  256, 0, stream>>>(xbf, wqkvT, qkv_b,
                                                           qws, kws, vtws);
  attn_kernel          <<<dim3(16, 32),  256, 0, stream>>>(qws, kws, vtws, aws);
  oproj_kernel         <<<dim3(32, 8),   256, 0, stream>>>(aws, owT, o_b,
                                                           (float*)d_out);
}
